// BQ_CorrBlock_67327907332136
// MI455X (gfx1250) — matrix operand order repaired and checked by reading, not compiled, not run on hardware
//
#include <hip/hip_runtime.h>

// Problem constants (fixed by the reference setup_inputs)
#define B_   4
#define NP   4096
#define DIM  128
#define CCH  64
#define NS   8
#define NG   8
#define R2   1.0f
#define CORR_SCALE 0.08838834764831845f   // 1/sqrt(128)
#define GN_EPS 1e-5f

typedef __attribute__((ext_vector_type(16))) _Float16 v16h;
typedef __attribute__((ext_vector_type(8)))  _Float16 v8h;
typedef __attribute__((ext_vector_type(8)))  float    v8f;

// ---------------------------------------------------------------------------
// K0: zero the GroupNorm accumulators (64 floats) -- must be re-zeroed per call
// ---------------------------------------------------------------------------
__global__ void k_zero(float* __restrict__ p, int n) {
    int i = blockIdx.x * blockDim.x + threadIdx.x;
    if (i < n) p[i] = 0.0f;
}

// ---------------------------------------------------------------------------
// K0b: convert out_w (64x64 f32) to f16 once per call -> WMMA A operand
// ---------------------------------------------------------------------------
__global__ void k_prep_w(const float* __restrict__ w, _Float16* __restrict__ wh, int n) {
    int i = blockIdx.x * blockDim.x + threadIdx.x;
    if (i < n) wh[i] = (_Float16)w[i];
}

// ---------------------------------------------------------------------------
// K1: fused ball-query + gathered-corr + grouped_xyz + 1x1 conv + GN partial
//     sums.  One thread per query point; whole xyz2[b] staged in LDS (48 KB).
//     Writes feat[b][4][NP][NS] to workspace; atomically accumulates
//     sum / sumsq per (b, group).
// ---------------------------------------------------------------------------
__global__ __launch_bounds__(256) void k_fused(
    const float* __restrict__ coords, const float* __restrict__ xyz2,
    const float* __restrict__ fmap1,  const float* __restrict__ fmap2,
    const float* __restrict__ conv_w, const float* __restrict__ conv_b,
    float* __restrict__ feat, float* __restrict__ statsAcc)
{
    const int b = blockIdx.y;
    const int n = blockIdx.x * 256 + threadIdx.x;

    __shared__ float sx[NP], sy[NP], sz[NP];
    __shared__ float s_sum[NG], s_ss[NG];

    // cooperative load of the whole point cloud for this batch
    for (int i = threadIdx.x; i < NP; i += 256) {
        const float* p = xyz2 + ((size_t)b * NP + i) * 3;
        sx[i] = p[0]; sy[i] = p[1]; sz[i] = p[2];
    }
    if (threadIdx.x < NG) { s_sum[threadIdx.x] = 0.0f; s_ss[threadIdx.x] = 0.0f; }
    __syncthreads();

    const float qx = coords[((size_t)b * NP + n) * 3 + 0];
    const float qy = coords[((size_t)b * NP + n) * 3 + 1];
    const float qz = coords[((size_t)b * NP + n) * 3 + 2];

    // ---- ball query: first NS indices with d2 < R2 (ascending order) ----
    int idx[NS];
    #pragma unroll
    for (int k = 0; k < NS; ++k) idx[k] = 0;
    int cnt = 0;
    for (int m = 0; m < NP; ++m) {
        float dx = qx - sx[m], dy = qy - sy[m], dz = qz - sz[m];
        float d2 = dx * dx + dy * dy + dz * dz;
        if (d2 < R2) {
            if      (cnt == 0) idx[0] = m;
            else if (cnt == 1) idx[1] = m;
            else if (cnt == 2) idx[2] = m;
            else if (cnt == 3) idx[3] = m;
            else if (cnt == 4) idx[4] = m;
            else if (cnt == 5) idx[5] = m;
            else if (cnt == 6) idx[6] = m;
            else               idx[7] = m;
            if (++cnt == NS) break;
        }
    }
    const int pad = (cnt > 0) ? idx[0] : 0;
    #pragma unroll
    for (int k = 0; k < NS; ++k) if (k >= cnt) idx[k] = pad;

    // ---- grouped_xyz = xyz2[idx] - coords ----
    float gx[NS], gy[NS], gz[NS];
    #pragma unroll
    for (int k = 0; k < NS; ++k) {
        int m = idx[k];
        gx[k] = sx[m] - qx; gy[k] = sy[m] - qy; gz[k] = sz[m] - qz;
    }

    // ---- gathered correlation: dot(fmap1[:,n], fmap2[:,idx_k]) ----
    float corr[NS];
    #pragma unroll
    for (int k = 0; k < NS; ++k) corr[k] = 0.0f;
    const float* f1  = fmap1 + (size_t)b * DIM * NP + n;
    const float* f2b = fmap2 + (size_t)b * DIM * NP;
    for (int d = 0; d < DIM; ++d) {
        float a = f1[(size_t)d * NP];                 // coalesced across threads
        const float* row = f2b + (size_t)d * NP;      // L2-resident gather row
        #pragma unroll
        for (int k = 0; k < NS; ++k)
            corr[k] = fmaf(a, row[idx[k]], corr[k]);
    }
    #pragma unroll
    for (int k = 0; k < NS; ++k) corr[k] *= CORR_SCALE;

    // ---- store feat[b][ch][n][s] ----
    {
        const size_t chs = (size_t)NP * NS;
        float* fb = feat + (size_t)b * 4 * chs + (size_t)n * NS;
        #pragma unroll
        for (int k = 0; k < NS; ++k) {
            fb[0 * chs + k] = corr[k];
            fb[1 * chs + k] = gx[k];
            fb[2 * chs + k] = gy[k];
            fb[3 * chs + k] = gz[k];
        }
    }

    // ---- 1x1 conv on the fly; accumulate GN sums per group ----
    float gsum[NG], gss[NG];
    #pragma unroll
    for (int g = 0; g < NG; ++g) { gsum[g] = 0.0f; gss[g] = 0.0f; }
    for (int c = 0; c < CCH; ++c) {
        float w0 = conv_w[c * 4 + 0], w1 = conv_w[c * 4 + 1];
        float w2 = conv_w[c * 4 + 2], w3 = conv_w[c * 4 + 3];
        float bb = conv_b[c];
        int g = c >> 3;
        #pragma unroll
        for (int s = 0; s < NS; ++s) {
            float v = fmaf(w0, corr[s],
                      fmaf(w1, gx[s],
                      fmaf(w2, gy[s],
                      fmaf(w3, gz[s], bb))));
            gsum[g] += v;
            gss[g]  = fmaf(v, v, gss[g]);
        }
    }
    #pragma unroll
    for (int g = 0; g < NG; ++g) {
        atomicAdd(&s_sum[g], gsum[g]);
        atomicAdd(&s_ss[g],  gss[g]);
    }
    __syncthreads();
    if (threadIdx.x < NG) {
        int g = threadIdx.x;
        atomicAdd(&statsAcc[(b * NG + g) * 2 + 0], s_sum[g]);
        atomicAdd(&statsAcc[(b * NG + g) * 2 + 1], s_ss[g]);
    }
}

// ---------------------------------------------------------------------------
// K2: finalize mean / rstd per (batch, group)
// ---------------------------------------------------------------------------
__global__ void k_finalize(const float* __restrict__ acc, float* __restrict__ mr) {
    int i = threadIdx.x;
    if (i >= B_ * NG) return;
    const float Ninv = 1.0f / ((float)(CCH / NG) * NP * NS);   // 1/262144
    float mu  = acc[i * 2 + 0] * Ninv;
    float var = acc[i * 2 + 1] * Ninv - mu * mu;
    mr[i * 2 + 0] = mu;
    mr[i * 2 + 1] = rsqrtf(var + GN_EPS);
}

// ---------------------------------------------------------------------------
// K3: re-apply conv (4 MACs -- cheaper than storing the 32MB post-conv x),
//     GroupNorm, PReLU, max over neighbors.
//     Output stored TRANSPOSED as y_t[b][n][c] in f16 so the WMMA B operand
//     becomes contiguous 16-byte chunks (global_load_b128 instead of 16x u16).
// ---------------------------------------------------------------------------
__global__ __launch_bounds__(256) void k_norm_max(
    const float* __restrict__ feat, const float* __restrict__ mr,
    const float* __restrict__ conv_w, const float* __restrict__ conv_b,
    const float* __restrict__ gamma, const float* __restrict__ beta,
    const float* __restrict__ prelu_a, _Float16* __restrict__ yt)
{
    const int b  = blockIdx.y;
    const int n0 = blockIdx.x * 64;

    __shared__ float sf[4][64 * NS];
    const size_t chs = (size_t)NP * NS;
    for (int i = threadIdx.x; i < 4 * 64 * NS; i += 256) {
        int ch = i / (64 * NS);
        int r  = i % (64 * NS);
        sf[ch][r] = feat[(size_t)b * 4 * chs + (size_t)ch * chs + (size_t)n0 * NS + r];
    }
    __syncthreads();

    const int c = threadIdx.x & 63;
    const float w0 = conv_w[c * 4 + 0], w1 = conv_w[c * 4 + 1];
    const float w2 = conv_w[c * 4 + 2], w3 = conv_w[c * 4 + 3];
    const float bb = conv_b[c];
    const int   g  = c >> 3;
    const float mu   = mr[(b * NG + g) * 2 + 0];
    const float rstd = mr[(b * NG + g) * 2 + 1];
    const float ga = gamma[c], be = beta[c];
    const float alpha = prelu_a[0];

    for (int ln = (threadIdx.x >> 6); ln < 64; ln += 4) {
        float mx = -3.4e38f;
        #pragma unroll
        for (int s = 0; s < NS; ++s) {
            float v = fmaf(w0, sf[0][ln * NS + s],
                      fmaf(w1, sf[1][ln * NS + s],
                      fmaf(w2, sf[2][ln * NS + s],
                      fmaf(w3, sf[3][ln * NS + s], bb))));
            v = (v - mu) * rstd * ga + be;
            v = (v >= 0.0f) ? v : alpha * v;
            mx = fmaxf(mx, v);
        }
        // transposed store: consecutive threads -> consecutive c (coalesced)
        yt[((size_t)b * NP + n0 + ln) * CCH + c] = (_Float16)mx;
    }
}

// ---------------------------------------------------------------------------
// K4: output projection out = out_w @ y + out_b via WMMA f16->f32.
//     Each wave: one 16(o) x 16(n) f32 tile, K=64 -> 2x v_wmma_f32_16x16x32_f16.
//     All fragments load as 16-byte global_load_b128 thanks to the f16 copies:
//       A: ow_h[o][c]   (row-major, K runs of 8 contiguous)
//       B: y_t[n][c]    (c contiguous = K contiguous per lane)
// ---------------------------------------------------------------------------
__global__ __launch_bounds__(256) void k_out_gemm(
    const _Float16* __restrict__ yt, const _Float16* __restrict__ ow_h,
    const float* __restrict__ out_b, float* __restrict__ out)
{
    const int b    = blockIdx.y;
    const int wave = threadIdx.x >> 5;
    const int lane = threadIdx.x & 31;
    const int o0   = (wave & 3) * 16;
    const int n0   = blockIdx.x * 32 + (wave >> 2) * 16;
    const int l15  = lane & 15;
    const int hi   = lane >> 4;           // which half of the wave

    const _Float16* arow = ow_h + (size_t)(o0 + l15) * CCH;
    const _Float16* brow = yt + ((size_t)b * NP + n0 + l15) * CCH;

    v8f acc = {};
    #pragma unroll
    for (int k0 = 0; k0 < CCH; k0 += 32) {
        // A fragment (16x32 f16, M=o, K=c): elem e -> K = (e&7) + ((e>>3)<<4) + hi*8
        //   = two contiguous runs of 8 halves: [k0+hi*8, +8) and [k0+16+hi*8, +8)
        v8h alo = *(const v8h*)(arow + k0 + hi * 8);
        v8h ahi = *(const v8h*)(arow + k0 + 16 + hi * 8);
        // B fragment (32x16 f16, K=c, N=n): elem e -> K = (e&7) + ((e>>3)<<4) + hi*8, N = l15
        //   = two contiguous runs of 8 halves: [k0+hi*8, +8) and [k0+16+hi*8, +8)
        v8h blo = *(const v8h*)(brow + k0 + hi * 8);
        v8h bhi = *(const v8h*)(brow + k0 + 16 + hi * 8);

        v16h a  = __builtin_shufflevector(alo, ahi, 0,1,2,3,4,5,6,7,8,9,10,11,12,13,14,15);
        v16h bm = __builtin_shufflevector(blo, bhi, 0,1,2,3,4,5,6,7,8,9,10,11,12,13,14,15);

        acc = __builtin_amdgcn_wmma_f32_16x16x32_f16(
            /*neg_a=*/false, a, /*neg_b=*/false, bm,
            /*c_mod=*/(short)0, acc, /*reuse_a=*/false, /*reuse_b=*/false);
    }
    // C/D layout: VGPR r -> row o0 + r + hi*8, col n0 + l15
    #pragma unroll
    for (int r = 0; r < 8; ++r) {
        int o = o0 + r + hi * 8;
        out[((size_t)b * CCH + o) * NP + n0 + l15] = acc[r] + out_b[o];
    }
}

// ---------------------------------------------------------------------------
extern "C" void kernel_launch(void* const* d_in, const int* in_sizes, int n_in,
                              void* d_out, int out_size, void* d_ws, size_t ws_size,
                              hipStream_t stream)
{
    const float* coords  = (const float*)d_in[0];
    const float* xyz2    = (const float*)d_in[1];
    const float* fmap1   = (const float*)d_in[2];
    const float* fmap2   = (const float*)d_in[3];
    const float* conv_w  = (const float*)d_in[4];
    const float* conv_b  = (const float*)d_in[5];
    const float* gamma   = (const float*)d_in[6];
    const float* beta    = (const float*)d_in[7];
    const float* prelu_a = (const float*)d_in[8];
    const float* out_w   = (const float*)d_in[9];
    const float* out_b   = (const float*)d_in[10];
    float* out = (float*)d_out;

    // workspace layout (all 16B-aligned)
    char* ws = (char*)d_ws;
    float*    feat     = (float*)ws;                                  // B*4*NP*NS f32   (2 MB)
    float*    statsAcc = feat + (size_t)B_ * 4 * NP * NS;             // B*NG*2 f32
    float*    meanRstd = statsAcc + B_ * NG * 2;                      // B*NG*2 f32
    _Float16* yt       = (_Float16*)(meanRstd + B_ * NG * 2);         // B*NP*CCH f16    (2 MB)
    _Float16* ow_h     = yt + (size_t)B_ * NP * CCH;                  // CCH*CCH f16     (8 KB)

    // K0: zero GN accumulators + f16 copy of out_w (every call; deterministic)
    k_zero<<<1, 64, 0, stream>>>(statsAcc, B_ * NG * 2);
    k_prep_w<<<(CCH * CCH + 255) / 256, 256, 0, stream>>>(out_w, ow_h, CCH * CCH);

    // K1: ball query + corr gather + feat + conv + GN partial sums
    k_fused<<<dim3(NP / 256, B_), 256, 0, stream>>>(
        coords, xyz2, fmap1, fmap2, conv_w, conv_b, feat, statsAcc);

    // K2: finalize mean / rstd
    k_finalize<<<1, 32, 0, stream>>>(statsAcc, meanRstd);

    // K3: normalize + PReLU + max over neighbors -> transposed f16 y
    k_norm_max<<<dim3(NP / 64, B_), 256, 0, stream>>>(
        feat, meanRstd, conv_w, conv_b, gamma, beta, prelu_a, yt);

    // K4: out projection with WMMA (all operands via global_load_b128)
    k_out_gemm<<<dim3(NP / 32, B_), 256, 0, stream>>>(yt, ow_h, out_b, out);
}
